// GCN_26963804684652
// MI455X (gfx1250) — matrix operand order repaired and checked by reading, not compiled, not run on hardware
//
#include <hip/hip_runtime.h>

// ---------------------------------------------------------------------------
// CDNA5 (gfx1250) GCN forward. wave32. WMMA f16->f32 for the dense GEMMs.
// B matrices pre-swizzled into per-lane fragment layout; edge norms
// precomputed once; all row buffers padded to 16 rows so WMMA epilogues are
// branch-free.
// ---------------------------------------------------------------------------

typedef __attribute__((ext_vector_type(16))) _Float16 v16h;
typedef __attribute__((ext_vector_type(4)))  _Float16 v4h;
typedef __attribute__((ext_vector_type(8)))  float    v8f;
typedef __attribute__((ext_vector_type(4)))  float    v4f;

// ---- A fragment: 16x32 f16 (MxK) tile, row-major, row stride ldk ----------
// A layout (contiguous K, matches packed B): lanes 0-15 -> M=lane, K 0..15;
// lanes 16-31 -> M=lane-16, K 16..31.
// Each lane's data = two contiguous 16B chunks -> global_load_b128 x2.
__device__ __forceinline__ v16h load_a_frag(const _Float16* __restrict__ A,
                                            int lane, int ldk) {
  int row = lane & 15;
  int kb  = (lane >> 4) << 4;            // 0 or 16
  const _Float16* p = A + (size_t)row * ldk;
  v16h a;
#pragma unroll
  for (int j = 0; j < 8; ++j) {
    a[j]     = p[kb + j];
    a[8 + j] = p[kb + 8 + j];
  }
  return a;
}

// ---------------------------------------------------------------------------
__global__ void k_f32_to_f16_v4(const float* __restrict__ in,
                                _Float16* __restrict__ out, int n4) {
  int i = blockIdx.x * blockDim.x + threadIdx.x;
  if (i < n4) {
    v4f v = ((const v4f*)in)[i];
    v4h h;
#pragma unroll
    for (int j = 0; j < 4; ++j) h[j] = (_Float16)v[j];
    ((v4h*)out)[i] = h;
  }
}

__global__ void k_zero_f32(float* __restrict__ p, int n) {
  int i = blockIdx.x * blockDim.x + threadIdx.x;
  if (i < n) p[i] = 0.0f;
}

// Pack W1 [128,32] f32 into 8 WMMA B fragments (kstep 0..3 x coltile 0..1).
// Fragment layout per lane l: col = l&15, kb = (l>>4)*16; element e -> K=kb+e.
__global__ void k_pack_b1(const float* __restrict__ W1, _Float16* __restrict__ W1p) {
  int idx = blockIdx.x * blockDim.x + threadIdx.x;   // 8*32*16 = 4096
  if (idx >= 4096) return;
  int e    = idx & 15;
  int l    = (idx >> 4) & 31;
  int frag = idx >> 9;
  int ks   = frag >> 1;
  int ct   = frag & 1;
  int col  = l & 15;
  int kb   = (l >> 4) << 4;
  W1p[idx] = (_Float16)W1[(size_t)(ks * 32 + kb + e) * 32 + ct * 16 + col];
}

// Pack W2 [32,16] f32 into 1 WMMA B fragment.
__global__ void k_pack_b2(const float* __restrict__ W2, _Float16* __restrict__ W2p) {
  int idx = blockIdx.x * blockDim.x + threadIdx.x;   // 32*16 = 512
  if (idx >= 512) return;
  int e   = idx & 15;
  int l   = idx >> 4;
  int col = l & 15;
  int kb  = (l >> 4) << 4;
  W2p[idx] = (_Float16)W2[(size_t)(kb + e) * 16 + col];
}

// deg[dst] += w  (self-loop +1 added later)
__global__ void k_degree(const int* __restrict__ dst,
                         const float* __restrict__ w,
                         float* __restrict__ deg, int E) {
  int e = blockIdx.x * blockDim.x + threadIdx.x;
  if (e < E) atomicAdd(&deg[dst[e]], w[e]);
}

__global__ void k_dinv(const float* __restrict__ deg,
                       float* __restrict__ dinv,
                       float* __restrict__ dginv, int n) {
  int i = blockIdx.x * blockDim.x + threadIdx.x;
  if (i < n) {
    float d  = deg[i] + 1.0f;
    dinv[i]  = rsqrtf(d);
    dginv[i] = 1.0f / d;
  }
}

// norm[e] = dinv[src]*w*dinv[dst], computed once, reused by both agg passes
__global__ void k_norm(const int* __restrict__ src, const int* __restrict__ dst,
                       const float* __restrict__ w, const float* __restrict__ dinv,
                       float* __restrict__ norm, int E) {
  int e = blockIdx.x * blockDim.x + threadIdx.x;
  if (e < E) norm[e] = dinv[src[e]] * w[e] * dinv[dst[e]];
}

// ---- GEMM1: h1[Np,32] = xh[Np,128] @ W1 (pre-packed fragments) ------------
__global__ void k_gemm1_wmma(const _Float16* __restrict__ xh,
                             const _Float16* __restrict__ W1p,
                             float* __restrict__ h1, int ntiles) {
  int gwave = (int)((blockIdx.x * blockDim.x + threadIdx.x) >> 5);
  int lane  = threadIdx.x & 31;
  if (gwave >= ntiles) return;                     // wave-uniform: EXEC stays full
  const _Float16* Arow = xh + (size_t)gwave * 16 * 128;
  const v16h* Bf = (const v16h*)W1p;
  v8f c0 = {}; v8f c1 = {};
#pragma unroll
  for (int ks = 0; ks < 4; ++ks) {
    v16h a  = load_a_frag(Arow + ks * 32, lane, 128);
    v16h b0 = Bf[(ks * 2 + 0) * 32 + lane];        // cols 0..15 fragment
    v16h b1 = Bf[(ks * 2 + 1) * 32 + lane];        // cols 16..31 fragment
    c0 = __builtin_amdgcn_wmma_f32_16x16x32_f16(false, a, false, b0, (short)0, c0, false, false);
    c1 = __builtin_amdgcn_wmma_f32_16x16x32_f16(false, a, false, b1, (short)0, c1, false, false);
  }
  // C layout: VGPR r -> M = r + 8*(lane/16); N = lane%16. Buffers are
  // 16-row padded, so stores are unconditional (branch-free epilogue).
  int col = lane & 15;
  int rb  = (lane >> 4) << 3;
  float* outp = h1 + ((size_t)(gwave * 16 + rb)) * 32 + col;
#pragma unroll
  for (int r = 0; r < 8; ++r) {
    outp[(size_t)r * 32]      = c0[r];
    outp[(size_t)r * 32 + 16] = c1[r];
  }
}

// ---- GEMM2: h2[Np,16] = hh[Np,32] @ W2 (pre-packed fragment) --------------
__global__ void k_gemm2_wmma(const _Float16* __restrict__ hh,
                             const _Float16* __restrict__ W2p,
                             float* __restrict__ h2, int ntiles) {
  int gwave = (int)((blockIdx.x * blockDim.x + threadIdx.x) >> 5);
  int lane  = threadIdx.x & 31;
  if (gwave >= ntiles) return;
  v16h a = load_a_frag(hh + (size_t)gwave * 16 * 32, lane, 32);
  v16h b = ((const v16h*)W2p)[lane];
  v8f c = {};
  c = __builtin_amdgcn_wmma_f32_16x16x32_f16(false, a, false, b, (short)0, c, false, false);
  int col = lane & 15;
  int rb  = (lane >> 4) << 3;
  float* outp = h2 + ((size_t)(gwave * 16 + rb)) * 16 + col;
#pragma unroll
  for (int r = 0; r < 8; ++r) outp[(size_t)r * 16] = c[r];
}

// ---- Edge scatter, layer 1: 32 lanes per edge (lane = feature) ------------
__global__ void k_agg1(const int* __restrict__ src, const int* __restrict__ dst,
                       const float* __restrict__ norm,
                       const float* __restrict__ h1, float* __restrict__ agg,
                       long long E) {
  long long tid = (long long)blockIdx.x * blockDim.x + threadIdx.x;
  if (tid >= E * 32) return;
  int       f = (int)(tid & 31);
  long long e = tid >> 5;
  int s = src[e], d = dst[e];
  atomicAdd(&agg[(size_t)d * 32 + f], norm[e] * h1[(size_t)s * 32 + f]);
}

// x1 = agg1 + h1/deg + b1; h = relu(x1); store f32 + f16 copies
__global__ void k_relu(const float* __restrict__ agg1, const float* __restrict__ h1,
                       const float* __restrict__ dginv, const float* __restrict__ b1,
                       float* __restrict__ hf, _Float16* __restrict__ hh, int n32) {
  int i = blockIdx.x * blockDim.x + threadIdx.x;
  if (i >= n32) return;
  int nidx = i >> 5, f = i & 31;
  float v = agg1[i] + h1[i] * dginv[nidx] + b1[f];
  v = fmaxf(v, 0.0f);
  hf[i] = v;
  hh[i] = (_Float16)v;
}

// ---- Edge scatter, layer 2: 16 lanes per edge -----------------------------
__global__ void k_agg2(const int* __restrict__ src, const int* __restrict__ dst,
                       const float* __restrict__ norm,
                       const float* __restrict__ h2, float* __restrict__ agg,
                       long long E) {
  long long tid = (long long)blockIdx.x * blockDim.x + threadIdx.x;
  if (tid >= E * 16) return;
  int       f = (int)(tid & 15);
  long long e = tid >> 4;
  int s = src[e], d = dst[e];
  atomicAdd(&agg[(size_t)d * 16 + f], norm[e] * h2[(size_t)s * 16 + f]);
}

// ---- Heads: x2 -> x1f -> sigmoid -> r; out = [r(0..N-1), x1f(N..2N-1)] ----
__global__ void k_final(const float* __restrict__ agg2, const float* __restrict__ h2,
                        const float* __restrict__ hf,   const float* __restrict__ dginv,
                        const float* __restrict__ b2,
                        const float* __restrict__ fc1w, const float* __restrict__ fc1b,
                        const float* __restrict__ fc2w, const float* __restrict__ fc2b,
                        float* __restrict__ out, int N) {
  int n = blockIdx.x * blockDim.x + threadIdx.x;
  if (n >= N) return;
  float dg  = dginv[n];
  float x1f = fc1b[0];
#pragma unroll
  for (int f = 0; f < 16; ++f) {
    float x2 = agg2[(size_t)n * 16 + f] + h2[(size_t)n * 16 + f] * dg + b2[f];
    x1f += x2 * fc1w[f];
  }
  float c = 1.0f / (1.0f + __expf(-x1f));
  float r = fc2b[0] + c * fc2w[32];
#pragma unroll
  for (int f = 0; f < 32; ++f) r += hf[(size_t)n * 32 + f] * fc2w[f];
  out[n]     = r;
  out[N + n] = x1f;
}

// ---------------------------------------------------------------------------
extern "C" void kernel_launch(void* const* d_in, const int* in_sizes, int n_in,
                              void* d_out, int out_size, void* d_ws, size_t ws_size,
                              hipStream_t stream) {
  const float* x    = (const float*)d_in[0];
  const int*   ei   = (const int*)  d_in[1];
  const float* ew   = (const float*)d_in[2];
  const float* W1   = (const float*)d_in[3];
  const float* b1   = (const float*)d_in[4];
  const float* W2   = (const float*)d_in[5];
  const float* b2   = (const float*)d_in[6];
  const float* fc1w = (const float*)d_in[7];
  const float* fc1b = (const float*)d_in[8];
  const float* fc2w = (const float*)d_in[9];
  const float* fc2b = (const float*)d_in[10];
  float* out = (float*)d_out;

  const int N = in_sizes[0] / 128;           // 100000
  const int E = in_sizes[2];                 // 3200000
  const int Np = (N + 15) & ~15;             // pad to 16-row tiles
  const int ntiles = Np / 16;
  const int* src = ei;                       // edge_index[0,:]
  const int* dst = ei + E;                   // edge_index[1,:]

  // --- bump allocator over d_ws (256B aligned) ---
  char* base = (char*)d_ws;
  size_t off = 0;
  auto alloc = [&](size_t bytes) -> char* {
    char* p = base + off;
    off += (bytes + 255) & ~(size_t)255;
    return p;
  };
  float*    h1    = (float*)   alloc((size_t)Np * 32 * 4);
  float*    agg1  = (float*)   alloc((size_t)Np * 32 * 4);
  float*    hf    = (float*)   alloc((size_t)Np * 32 * 4);
  float*    deg   = (float*)   alloc((size_t)N * 4);
  float*    dinv  = (float*)   alloc((size_t)N * 4);
  float*    dginv = (float*)   alloc((size_t)N * 4);
  float*    norm  = (float*)   alloc((size_t)E * 4);
  _Float16* W1p   = (_Float16*)alloc(8  * 32 * 16 * 2);   // 8 packed fragments
  _Float16* W2p   = (_Float16*)alloc(1  * 32 * 16 * 2);   // 1 packed fragment
  // union region: xh (f16, dead after GEMM1) aliases {hh, h2, agg2}
  char* uni = alloc((size_t)Np * 128 * 2);
  _Float16* xh   = (_Float16*)uni;
  _Float16* hh   = (_Float16*)uni;
  float*    h2   = (float*)(uni + (((size_t)Np * 32 * 2 + 255) & ~(size_t)255));
  float*    agg2 = (float*)((char*)h2 + (((size_t)Np * 16 * 4 + 255) & ~(size_t)255));
  (void)ws_size; (void)n_in; (void)out_size;

  const int B = 256;
  auto nb = [](long long n, int b) { return (unsigned)((n + b - 1) / b); };

  // 1) f16 copy of x (vectorized) + pre-swizzled B fragments for W1/W2
  k_f32_to_f16_v4<<<nb((long long)N * 32, B), B, 0, stream>>>(x, xh, N * 32);
  k_pack_b1<<<nb(4096, B), B, 0, stream>>>(W1, W1p);
  k_pack_b2<<<nb(512, B), B, 0, stream>>>(W2, W2p);

  // 2) degrees -> dinv, 1/deg -> per-edge norms
  k_zero_f32<<<nb(N, B), B, 0, stream>>>(deg, N);
  k_degree<<<nb(E, B), B, 0, stream>>>(dst, ew, deg, E);
  k_dinv<<<nb(N, B), B, 0, stream>>>(deg, dinv, dginv, N);
  k_norm<<<nb(E, B), B, 0, stream>>>(src, dst, ew, dinv, norm, E);

  // 3) GEMM1 (WMMA): h1 = xh @ W1
  k_gemm1_wmma<<<nb((long long)ntiles * 32, B), B, 0, stream>>>(xh, W1p, h1, ntiles);

  // 4) layer-1 aggregation (xh now dead)
  k_zero_f32<<<nb((long long)N * 32, B), B, 0, stream>>>(agg1, N * 32);
  k_agg1<<<nb((long long)E * 32, B), B, 0, stream>>>(src, dst, norm, h1, agg1, E);

  // 5) x1 -> relu -> h (f32 + f16)
  k_relu<<<nb((long long)N * 32, B), B, 0, stream>>>(agg1, h1, dginv, b1, hf, hh, N * 32);

  // 6) GEMM2 (WMMA): h2 = hh @ W2
  k_gemm2_wmma<<<nb((long long)ntiles * 32, B), B, 0, stream>>>(hh, W2p, h2, ntiles);

  // 7) layer-2 aggregation
  k_zero_f32<<<nb((long long)N * 16, B), B, 0, stream>>>(agg2, N * 16);
  k_agg2<<<nb((long long)E * 16, B), B, 0, stream>>>(src, dst, norm, h2, agg2, E);

  // 8) heads + outputs
  k_final<<<nb(N, B), B, 0, stream>>>(agg2, h2, hf, dginv, b2,
                                      fc1w, fc1b, fc2w, fc2b, out, N);
}
